// Conv_Adapter_v13_32933809225989
// MI455X (gfx1250) — matrix operand order repaired and checked by reading, not compiled, not run on hardware
//
#include <hip/hip_runtime.h>
#include <hip/hip_bf16.h>

// ---------------- types / constants ----------------
typedef __attribute__((ext_vector_type(16))) _Float16 v16h;
typedef __attribute__((ext_vector_type(8)))  _Float16 v8h;
typedef __attribute__((ext_vector_type(8)))  float    v8f;

#define TOKN 576
#define NB   64
#define DIM  1024
#define TXT  512
#define NTOK 1000
#define NPAD 1024
#define HH   24
#define QSCALE 0.04419417382415922f   // 512^-0.5

// ---------------- helpers ----------------
__device__ __forceinline__ v8f vzero8() {
  v8f z;
  #pragma unroll
  for (int i = 0; i < 8; ++i) z[i] = 0.f;
  return z;
}

// A fragment (16xK tile, 16-bit): lane<16 holds row (lane&15), K {0..7,16..23};
// lane>=16 same row, K {8..15,24..31}.  Source row-major f16, ld in elements.
__device__ __forceinline__ v16h ldfragA(const _Float16* base, int ld) {
  int lane = threadIdx.x & 31;
  const _Float16* p = base + (size_t)(lane & 15) * ld + ((lane >> 4) << 3);
  union { v16h v; v8h h[2]; } f;
  f.h[0] = *(const v8h*)p;
  f.h[1] = *(const v8h*)(p + 16);
  return f.v;
}
// B fragment (Kx16 tile, 16-bit), loaded from B^T row-major (row = N col, contiguous K):
// lane<16 = col (lane&15), K {0..7,16..23} ; lane>=16 same col, K {8..15,24..31}.
__device__ __forceinline__ v16h ldfragB(const _Float16* base, int ld) {
  int lane = threadIdx.x & 31;
  const _Float16* p = base + (size_t)(lane & 15) * ld + ((lane >> 4) << 3);
  union { v16h v; v8h h[2]; } f;
  f.h[0] = *(const v8h*)p;
  f.h[1] = *(const v8h*)(p + 16);
  return f.v;
}
__device__ __forceinline__ v8f wmma_f16(v16h a, v16h b, v8f c) {
  return __builtin_amdgcn_wmma_f32_16x16x32_f16(false, a, false, b, (short)0, c,
                                                false, false);
}
__device__ __forceinline__ float redsum32(float v) {
  #pragma unroll
  for (int m = 16; m >= 1; m >>= 1) v += __shfl_xor(v, m, 32);
  return v;
}
__device__ __forceinline__ float redmax32(float v) {
  #pragma unroll
  for (int m = 16; m >= 1; m >>= 1) v = fmaxf(v, __shfl_xor(v, m, 32));
  return v;
}
__device__ __forceinline__ float redsum16(float v) {
  #pragma unroll
  for (int m = 8; m >= 1; m >>= 1) v += __shfl_xor(v, m, 16);
  return v;
}
__device__ __forceinline__ float redmax16(float v) {
  #pragma unroll
  for (int m = 8; m >= 1; m >>= 1) v = fmaxf(v, __shfl_xor(v, m, 16));
  return v;
}

// ---------------- K1: kv layernorm -> f16 (padded rows zeroed) ----------------
__global__ void __launch_bounds__(256) k_kv_ln(const float* __restrict__ tf,
                                               const float* __restrict__ g,
                                               const float* __restrict__ bb,
                                               _Float16* __restrict__ kvn) {
  int wid  = blockIdx.x * 8 + (threadIdx.x >> 5);
  int lane = threadIdx.x & 31;
  int b = wid >> 10;        // / NPAD
  int n = wid & (NPAD - 1);
  _Float16* dst = kvn + ((size_t)b * NPAD + n) * TXT;
  if (n >= NTOK) {
    #pragma unroll
    for (int i = 0; i < 16; ++i) dst[lane + 32 * i] = (_Float16)0.f;
    return;
  }
  const float* src = tf + ((size_t)n * NB + b) * TXT;
  float v[16], s = 0.f, sq = 0.f;
  #pragma unroll
  for (int i = 0; i < 16; ++i) {
    float x = src[lane + 32 * i];
    v[i] = x; s += x; sq += x * x;
  }
  s = redsum32(s); sq = redsum32(sq);
  float mean = s * (1.f / TXT);
  float var  = sq * (1.f / TXT) - mean * mean;
  float rs   = rsqrtf(var + 1e-5f);
  #pragma unroll
  for (int i = 0; i < 16; ++i) {
    int e = lane + 32 * i;
    dst[e] = (_Float16)((v[i] - mean) * rs * g[e] + bb[e]);
  }
}

// ---------------- K1b: kv[b][n][e] -> kvt[b][e][n] (LDS-tiled) ----------------
__global__ void __launch_bounds__(256) k_kv_tr(const _Float16* __restrict__ kvn,
                                               _Float16* __restrict__ kvt) {
  __shared__ __align__(16) _Float16 tile[64][65];
  int blk = blockIdx.x;
  int b  = blk >> 7;             // 128 tiles per batch
  int r  = blk & 127;
  int n0 = (r >> 3) * 64;        // 16 n-tiles
  int e0 = (r & 7) * 64;         // 8 e-tiles
  int tid = threadIdx.x;
  #pragma unroll
  for (int i = 0; i < 16; ++i) {
    int idx = tid + 256 * i;
    int rr = idx >> 6, cc = idx & 63;
    tile[rr][cc] = kvn[((size_t)b * NPAD + n0 + rr) * TXT + e0 + cc];
  }
  __syncthreads();
  #pragma unroll
  for (int i = 0; i < 16; ++i) {
    int idx = tid + 256 * i;
    int rr = idx >> 6, cc = idx & 63;
    kvt[((size_t)b * TXT + e0 + rr) * NPAD + n0 + cc] = tile[cc][rr];
  }
}

// ---------------- K0a: Wq (1024x512 f32) -> WqT (512x1024 f16) ----------------
__global__ void __launch_bounds__(256) k_wq_tr(const float* __restrict__ wq,
                                               _Float16* __restrict__ wqt) {
  __shared__ float tile[32][33];
  int blk = blockIdx.x;
  int d0 = (blk >> 4) * 32;   // 32 d-tiles
  int e0 = (blk & 15) * 32;   // 16 e-tiles
  int tid = threadIdx.x;
  #pragma unroll
  for (int i = 0; i < 4; ++i) {
    int idx = tid + 256 * i;
    int rr = idx >> 5, cc = idx & 31;
    tile[rr][cc] = wq[(size_t)(d0 + rr) * TXT + e0 + cc];
  }
  __syncthreads();
  #pragma unroll
  for (int i = 0; i < 4; ++i) {
    int idx = tid + 256 * i;
    int rr = idx >> 5, cc = idx & 31;
    wqt[(size_t)(e0 + rr) * DIM + d0 + cc] = (_Float16)tile[cc][rr];
  }
}

// ---------------- K0b: f32 -> f16 elementwise ----------------
__global__ void __launch_bounds__(256) k_cvt_f16(const float* __restrict__ src,
                                                 _Float16* __restrict__ dst, int n) {
  int i = blockIdx.x * 256 + threadIdx.x;
  if (i < n) dst[i] = (_Float16)src[i];
}

// ---------------- K2: fused LayerNorm + q GEMM (q1 and q2) ----------------
__global__ void __launch_bounds__(256) k_qgemm(
    const float* __restrict__ x,
    const float* __restrict__ g1, const float* __restrict__ b1,
    const float* __restrict__ g2, const float* __restrict__ b2,
    const _Float16* __restrict__ wq1t, const _Float16* __restrict__ wq2t,
    _Float16* __restrict__ q1, _Float16* __restrict__ q2) {
  __shared__ __align__(16) _Float16 Xs[16 * DIM];  // 32 KB
  int blk = blockIdx.x;
  int b  = blk / 36;
  int t0 = (blk % 36) * 16;
  int tid = threadIdx.x, w = tid >> 5, lane = tid & 31;
  int hi8 = (lane >> 4) << 3;

  for (int pass = 0; pass < 2; ++pass) {
    const float* lg = pass ? g2 : g1;
    const float* lb = pass ? b2 : b1;
    const _Float16* wt = pass ? wq2t : wq1t;
    _Float16* qo = pass ? q2 : q1;
    // LayerNorm: 16 rows, 2 per wave
    for (int rr = 0; rr < 2; ++rr) {
      int row = w * 2 + rr;
      const float* src = x + ((size_t)(1 + t0 + row) * NB + b) * DIM;
      float v[32], s = 0.f, sq = 0.f;
      #pragma unroll
      for (int i = 0; i < 32; ++i) {
        float xx = src[lane + 32 * i];
        v[i] = xx; s += xx; sq += xx * xx;
      }
      s = redsum32(s); sq = redsum32(sq);
      float mean = s * (1.f / DIM);
      float var  = sq * (1.f / DIM) - mean * mean;
      float rs   = rsqrtf(var + 1e-5f);
      #pragma unroll
      for (int i = 0; i < 32; ++i) {
        int d = lane + 32 * i;
        Xs[(size_t)row * DIM + d] = (_Float16)((v[i] - mean) * rs * lg[d] + lb[d]);
      }
    }
    __syncthreads();
    // GEMM 16x512, K=1024; 4 e-tiles per wave
    v8f acc[4];
    #pragma unroll
    for (int j = 0; j < 4; ++j) acc[j] = vzero8();
    #pragma unroll 1
    for (int k = 0; k < 32; ++k) {
      int k0 = k * 32;
      v16h a = ldfragA(Xs + k0, DIM);
      #pragma unroll
      for (int j = 0; j < 4; ++j) {
        int e0 = (w * 4 + j) * 16;
        v16h bf = ldfragB(wt + (size_t)e0 * DIM + k0, DIM);
        acc[j] = wmma_f16(a, bf, acc[j]);
      }
    }
    _Float16* qb = qo + ((size_t)b * TOKN + t0) * TXT;
    #pragma unroll
    for (int j = 0; j < 4; ++j) {
      int e0 = (w * 4 + j) * 16;
      #pragma unroll
      for (int r = 0; r < 8; ++r)
        qb[(size_t)(r + hi8) * TXT + e0 + (lane & 15)] = (_Float16)acc[j][r];
    }
    __syncthreads();
  }
}

// ---------------- K3: attention (scores + softmax + P@V), one 16-row tile ----------------
__global__ void __launch_bounds__(256) k_attn(
    const _Float16* __restrict__ q, const _Float16* __restrict__ kvn,
    const _Float16* __restrict__ kvt, _Float16* __restrict__ o) {
  __shared__ __align__(16) _Float16 Ps[16 * NPAD];  // 32 KB
  __shared__ float rmaxp[8][16];
  __shared__ float rsump[8][16];
  __shared__ float rowmax[16];
  __shared__ float rowsuminv[16];
  int blk = blockIdx.x;
  int b  = blk / 36;
  int t0 = (blk % 36) * 16;
  int tid = threadIdx.x, w = tid >> 5, lane = tid & 31;
  int hi8 = (lane >> 4) << 3;
  int nb = w * 128;   // this wave's key slice

  // phase 1: S = q @ kv^T   (K = 512)
  const _Float16* qb = q + ((size_t)b * TOKN + t0) * TXT;
  v8f S[8];
  #pragma unroll
  for (int j = 0; j < 8; ++j) S[j] = vzero8();
  #pragma unroll 1
  for (int k = 0; k < 16; ++k) {
    int k0 = k * 32;
    v16h a = ldfragA(qb + k0, TXT);
    #pragma unroll
    for (int j = 0; j < 8; ++j) {
      v16h bf = ldfragB(kvn + ((size_t)b * NPAD + nb + j * 16) * TXT + k0, TXT);
      S[j] = wmma_f16(a, bf, S[j]);
    }
  }
  // scale + mask + partial row max
  float pm[8];
  #pragma unroll
  for (int r = 0; r < 8; ++r) pm[r] = -3.0e38f;
  #pragma unroll
  for (int j = 0; j < 8; ++j) {
    int n = nb + j * 16 + (lane & 15);
    bool ok = (n < NTOK);
    #pragma unroll
    for (int r = 0; r < 8; ++r) {
      float s = ok ? S[j][r] * QSCALE : -3.0e38f;
      S[j][r] = s;
      pm[r] = fmaxf(pm[r], s);
    }
  }
  #pragma unroll
  for (int r = 0; r < 8; ++r) {
    float m = redmax16(pm[r]);
    if ((lane & 15) == 0) rmaxp[w][r + hi8] = m;
  }
  __syncthreads();
  if (tid < 16) {
    float m = -3.0e38f;
    for (int ww = 0; ww < 8; ++ww) m = fmaxf(m, rmaxp[ww][tid]);
    rowmax[tid] = m;
  }
  __syncthreads();
  // exp, partial sums, stash P (f16) in LDS
  float ps[8];
  #pragma unroll
  for (int r = 0; r < 8; ++r) {
    float mrow = rowmax[r + hi8];
    float s = 0.f;
    #pragma unroll
    for (int j = 0; j < 8; ++j) {
      float p = __expf(S[j][r] - mrow);
      S[j][r] = p;
      s += p;
    }
    ps[r] = s;
  }
  #pragma unroll
  for (int r = 0; r < 8; ++r) {
    float s = redsum16(ps[r]);
    if ((lane & 15) == 0) rsump[w][r + hi8] = s;
  }
  #pragma unroll
  for (int j = 0; j < 8; ++j)
    #pragma unroll
    for (int r = 0; r < 8; ++r)
      Ps[(size_t)(r + hi8) * NPAD + nb + j * 16 + (lane & 15)] = (_Float16)S[j][r];
  __syncthreads();
  if (tid < 16) {
    float s = 0.f;
    for (int ww = 0; ww < 8; ++ww) s += rsump[ww][tid];
    rowsuminv[tid] = 1.f / s;
  }
  __syncthreads();

  // phase 2: O = P @ kv   (K = 1024, 4 e-tiles per wave)
  v8f O[4];
  #pragma unroll
  for (int j = 0; j < 4; ++j) O[j] = vzero8();
  #pragma unroll 1
  for (int k = 0; k < 32; ++k) {
    int k0 = k * 32;
    v16h a = ldfragA(Ps + k0, NPAD);
    #pragma unroll
    for (int j = 0; j < 4; ++j) {
      int e0 = w * 64 + j * 16;
      v16h bf = ldfragB(kvt + ((size_t)b * TXT + e0) * NPAD + k0, NPAD);
      O[j] = wmma_f16(a, bf, O[j]);
    }
  }
  _Float16* ob = o + ((size_t)b * TOKN + t0) * TXT;
  #pragma unroll
  for (int r = 0; r < 8; ++r) {
    float ri = rowsuminv[r + hi8];
    #pragma unroll
    for (int j = 0; j < 4; ++j) {
      int e0 = w * 64 + j * 16;
      ob[(size_t)(r + hi8) * TXT + e0 + (lane & 15)] = (_Float16)(O[j][r] * ri);
    }
  }
}

// ---------------- K4: channel mean/max + 3x3 dwconv + sigmoid gates ----------------
__global__ void __launch_bounds__(256) k_gates(
    const _Float16* __restrict__ o1, const _Float16* __restrict__ o2,
    const float* __restrict__ wdw, float* __restrict__ g1, float* __restrict__ g2) {
  __shared__ float sa[TOKN];
  __shared__ float sm[TOKN];
  int b = blockIdx.x;
  int tid = threadIdx.x, w = tid >> 5, lane = tid & 31;
  for (int i = 0; i < TOKN / 8; ++i) {
    int t = w + 8 * i;
    const _Float16* p1 = o1 + ((size_t)b * TOKN + t) * TXT;
    const _Float16* p2 = o2 + ((size_t)b * TOKN + t) * TXT;
    float s = 0.f, m = -3.0e38f;
    #pragma unroll
    for (int j = 0; j < 16; ++j) {
      int c = lane + 32 * j;
      s += (float)p1[c];
      m = fmaxf(m, (float)p2[c]);
    }
    s = redsum32(s);
    m = redmax32(m);
    if (lane == 0) { sa[t] = s * (1.f / TXT); sm[t] = m; }
  }
  __syncthreads();
  for (int t = tid; t < TOKN; t += 256) {
    int hh = t / HH, cc0 = t % HH;
    float a1 = 0.f, a2 = 0.f;
    #pragma unroll
    for (int di = -1; di <= 1; ++di)
      #pragma unroll
      for (int dj = -1; dj <= 1; ++dj) {
        int rr = hh + di, cc = cc0 + dj;
        if (rr >= 0 && rr < HH && cc >= 0 && cc < HH) {
          float kk = wdw[(di + 1) * 3 + (dj + 1)];
          a1 += sa[rr * HH + cc] * kk;
          a2 += sm[rr * HH + cc] * kk;
        }
      }
    g1[(size_t)b * TOKN + t] = 1.f / (1.f + __expf(-a1));
    g2[(size_t)b * TOKN + t] = 1.f / (1.f + __expf(-a2));
  }
}

// ---------------- K5: gated combine (fused) + final 512->1024 GEMM ----------------
__global__ void __launch_bounds__(256) k_fgemm(
    const _Float16* __restrict__ o1, const _Float16* __restrict__ o2,
    const float* __restrict__ g1, const float* __restrict__ g2,
    const _Float16* __restrict__ wcf, float* __restrict__ out) {
  __shared__ __align__(16) _Float16 Xs[16 * TXT];  // 16 KB
  int blk = blockIdx.x;
  int b   = blk / (36 * 8);
  int rem = blk % (36 * 8);
  int t0  = (rem / 8) * 16;
  int ob0 = (rem % 8) * 128;
  int tid = threadIdx.x, w = tid >> 5, lane = tid & 31;
  int hi8 = (lane >> 4) << 3;
  {
    int r = tid >> 4;
    int t = t0 + r;
    float gg1 = g1[(size_t)b * TOKN + t];
    float gg2 = g2[(size_t)b * TOKN + t];
    const _Float16* p1 = o1 + ((size_t)b * TOKN + t) * TXT;
    const _Float16* p2 = o2 + ((size_t)b * TOKN + t) * TXT;
    #pragma unroll
    for (int i = 0; i < 32; ++i) {
      int c = (tid & 15) + 16 * i;
      Xs[(size_t)r * TXT + c] = (_Float16)((float)p1[c] * gg1 + (float)p2[c] * gg2);
    }
  }
  __syncthreads();
  int oc = ob0 + w * 16;
  v8f acc = vzero8();
  #pragma unroll 1
  for (int k = 0; k < 16; ++k) {
    int k0 = k * 32;
    v16h a  = ldfragA(Xs + k0, TXT);
    v16h bf = ldfragB(wcf + (size_t)oc * TXT + k0, TXT);  // conv_final rows = B^T rows
    acc = wmma_f16(a, bf, acc);
  }
  #pragma unroll
  for (int r = 0; r < 8; ++r)
    out[((size_t)(1 + t0 + r + hi8) * NB + b) * DIM + oc + (lane & 15)] = acc[r];
}

// ---------------- launcher ----------------
extern "C" void kernel_launch(void* const* d_in, const int* in_sizes, int n_in,
                              void* d_out, int out_size, void* d_ws, size_t ws_size,
                              hipStream_t stream) {
  (void)in_sizes; (void)n_in; (void)out_size; (void)ws_size;
  const float* x    = (const float*)d_in[0];
  const float* tf   = (const float*)d_in[1];
  const float* lq1g = (const float*)d_in[2];
  const float* lq1b = (const float*)d_in[3];
  const float* lq2g = (const float*)d_in[4];
  const float* lq2b = (const float*)d_in[5];
  const float* lkvg = (const float*)d_in[6];
  const float* lkvb = (const float*)d_in[7];
  const float* wq1  = (const float*)d_in[8];
  const float* wq2  = (const float*)d_in[9];
  const float* wdw  = (const float*)d_in[10];
  const float* wcf  = (const float*)d_in[11];
  float* out = (float*)d_out;

  char* ws = (char*)d_ws;
  size_t off = 0;
  auto alloc = [&](size_t bytes) {
    void* p = ws + off;
    off += (bytes + 255) & ~(size_t)255;
    return p;
  };
  _Float16* kvn  = (_Float16*)alloc((size_t)NB * NPAD * TXT * 2);   // 64 MB
  _Float16* kvt  = (_Float16*)alloc((size_t)NB * TXT * NPAD * 2);   // 64 MB
  _Float16* wq1t = (_Float16*)alloc((size_t)TXT * DIM * 2);
  _Float16* wq2t = (_Float16*)alloc((size_t)TXT * DIM * 2);
  _Float16* wcfh = (_Float16*)alloc((size_t)DIM * TXT * 2);
  _Float16* q1   = (_Float16*)alloc((size_t)NB * TOKN * TXT * 2);   // 36 MB
  _Float16* q2   = (_Float16*)alloc((size_t)NB * TOKN * TXT * 2);
  _Float16* o1   = (_Float16*)alloc((size_t)NB * TOKN * TXT * 2);
  _Float16* o2   = (_Float16*)alloc((size_t)NB * TOKN * TXT * 2);
  float*    gt1  = (float*)alloc((size_t)NB * TOKN * 4);
  float*    gt2  = (float*)alloc((size_t)NB * TOKN * 4);

  // cls passthrough row: out[0][b][d] = x[0][b][d]
  hipMemcpyAsync(out, x, (size_t)NB * DIM * sizeof(float),
                 hipMemcpyDeviceToDevice, stream);

  k_kv_ln <<<(NB * NPAD) / 8, 256, 0, stream>>>(tf, lkvg, lkvb, kvn);
  k_kv_tr <<<NB * 128, 256, 0, stream>>>(kvn, kvt);
  k_wq_tr <<<512, 256, 0, stream>>>(wq1, wq1t);
  k_wq_tr <<<512, 256, 0, stream>>>(wq2, wq2t);
  k_cvt_f16<<<(DIM * TXT + 255) / 256, 256, 0, stream>>>(wcf, wcfh, DIM * TXT);
  k_qgemm<<<NB * 36, 256, 0, stream>>>(x, lq1g, lq1b, lq2g, lq2b, wq1t, wq2t, q1, q2);
  k_attn <<<NB * 36, 256, 0, stream>>>(q1, kvn, kvt, o1);
  k_attn <<<NB * 36, 256, 0, stream>>>(q2, kvn, kvt, o2);
  k_gates<<<NB, 256, 0, stream>>>(o1, o2, wdw, gt1, gt2);
  k_fgemm<<<NB * 36 * 8, 256, 0, stream>>>(o1, o2, gt1, gt2, wcfh, out);
}
